// PGTAttention_3324304687864
// MI455X (gfx1250) — matrix operand order repaired and checked by reading, not compiled, not run on hardware
//
#include <hip/hip_runtime.h>
#include <hip/hip_bf16.h>
#include <cstddef>

// ---------------------------------------------------------------------------
// PGT attention block on gfx1250 (CDNA5, wave32, WMMA).
// B=2, S=2048, E=2048, H=8, D=256, layer_idx=3, eps=1e-5.
//   qkv  = hidden @ w_attn                      (bf16 WMMA, f32 acc)
//   attn = flash-softmax(Q K^T / (16*4)) V      (bf16 WMMA, f32 online softmax)
//   o    = PMSNorm_D(attn) * norm_w             (f32 VALU epilogue)
//   out  = o @ w_proj                           (bf16 WMMA, f32 acc -> f32 out)
//
// GEMMs: 128x128 block tile, K-step 32, 8 waves (4m x 2n), wave tile 32x64
// (8 WMMA / wave / K-step), double-buffered LDS -> 1 barrier per K-step with
// staging of tile k+1 overlapped with compute of tile k.
// Attention: K-tile staged with GLOBAL_LOAD_ASYNC_TO_LDS_B128 (ASYNCcnt path),
// V transposed via VGPRs.
// ---------------------------------------------------------------------------

typedef __bf16 bf16_t;
typedef __attribute__((ext_vector_type(16))) __bf16 v16bf;
typedef __attribute__((ext_vector_type(8)))  __bf16 v8bf;
typedef __attribute__((ext_vector_type(8)))  float  v8f;
typedef int v4i_gcc __attribute__((vector_size(16)));   // matches builtin proto

static constexpr int Bb   = 2;
static constexpr int Ss   = 2048;
static constexpr int Ee   = 2048;
static constexpr int Hh   = 8;
static constexpr int Dd   = 256;      // head dim == PMSNorm width
static constexpr int QKV_LD = 3 * Ee; // 6144
static constexpr float EPSf = 1e-5f;
static constexpr float NEGf = -3.4028235e38f;

#if __has_builtin(__builtin_amdgcn_global_load_async_to_lds_b128)
#define HAVE_ASYNC_LDS 1
#endif

__device__ inline void wait_async0() {
#if __has_builtin(__builtin_amdgcn_s_wait_asynccnt)
    __builtin_amdgcn_s_wait_asynccnt(0);
#else
    asm volatile("s_wait_asynccnt 0" ::: "memory");
#endif
}

__device__ inline v8f wmma_bf16(v16bf a, v16bf b, v8f c) {
    // D = A(16x32 bf16) * B(32x16 bf16) + C(16x16 f32)
    return __builtin_amdgcn_wmma_f32_16x16x32_bf16(
        /*neg_a=*/false, a, /*neg_b=*/false, b,
        /*c_mod=*/(short)0, c, /*reuse_a=*/false, /*reuse_b=*/false);
}

// A-matrix fragment (16x32, bf16), ISA layout:
//   lane L (row M = L&15):  halves 0..7  -> K = hioff..hioff+7
//                           halves 8..15 -> K = hioff+16..hioff+23
//   hioff = 8 for lanes 16..31, else 0.
__device__ inline v16bf load_a_frag(const bf16_t* p, int ld) {
    const int lane  = threadIdx.x & 31;
    const int row   = lane & 15;
    const int hioff = (lane & 16) ? 8 : 0;
    const bf16_t* r = p + row * ld + hioff;
    v8bf lo = *(const v8bf*)(r);
    v8bf hi = *(const v8bf*)(r + 16);
    v16bf v;
#pragma unroll
    for (int i = 0; i < 8; ++i) { v[i] = lo[i]; v[8 + i] = hi[i]; }
    return v;
}

// B-matrix fragment (32x16, bf16), ISA layout:
//   lane L (col N = L&15): halves 0..7 -> K = koff..koff+7,
//                          halves 8..15 -> K = koff+16..koff+23, koff = (L>=16 ? 8 : 0)
// Source stored "transposed" [N][K] with stride ld -> two 16B runs per lane, 32B apart.
__device__ inline v16bf load_b_frag(const bf16_t* p, int ld) {
    const int lane = threadIdx.x & 31;
    const int n    = lane & 15;
    const int koff = (lane & 16) ? 8 : 0;
    const bf16_t* r = p + n * ld + koff;
    v8bf lo = *(const v8bf*)(r);
    v8bf hi = *(const v8bf*)(r + 16);
    v16bf v;
#pragma unroll
    for (int i = 0; i < 8; ++i) { v[i] = lo[i]; v[8 + i] = hi[i]; }
    return v;
}

// ---------------------------------------------------------------------------
// Tiled GEMM: C[M,N] = A[M,K] * B[K,N], bf16 WMMA with f32 accumulation.
// 128x128 block, 256 threads = 8 waves (4m x 2n), wave tile 32x64,
// double-buffered LDS, one barrier per K-step.
// ---------------------------------------------------------------------------
template <typename TA, typename TO>
__global__ __launch_bounds__(256)
void gemm_bf16(const TA* __restrict__ A, const float* __restrict__ B,
               TO* __restrict__ C, int M, int N, int K) {
    __shared__ __align__(32) bf16_t sA[2][128 * 32];  // [row][k]
    __shared__ __align__(32) bf16_t sB[2][128 * 32];  // [col][k] (transposed)

    const int tid  = threadIdx.x;
    const int m0   = blockIdx.y * 128;
    const int n0   = blockIdx.x * 128;
    const int wave = tid >> 5;
    const int wm   = (wave & 3) * 32;   // wave row offset (4 m-waves)
    const int wn   = (wave >> 2) * 64;  // wave col offset (2 n-waves)

    v8f acc[2][4] = {};

    auto stage = [&](int k0, int buf) {
        // A tile 128x32 (convert to bf16), coalesced along k.
#pragma unroll
        for (int i = 0; i < 16; ++i) {
            int e = tid + i * 256;            // 0..4095
            int row = e >> 5, col = e & 31;
            sA[buf][row * 32 + col] =
                (bf16_t)(float)A[(size_t)(m0 + row) * K + (k0 + col)];
        }
        // B tile 32x128, stored transposed [col][k], coalesced along n.
#pragma unroll
        for (int i = 0; i < 16; ++i) {
            int e = tid + i * 256;            // 0..4095
            int row = e >> 7, col = e & 127;
            sB[buf][col * 32 + row] =
                (bf16_t)B[(size_t)(k0 + row) * N + (n0 + col)];
        }
        if (k0 + 32 < K) {  // pull next K slab toward L2 (global_prefetch_b8)
            __builtin_prefetch(&A[(size_t)(m0 + wave * 16) * K + k0 + 32], 0, 2);
            __builtin_prefetch(&B[(size_t)(k0 + 32) * N + n0 + (tid & 127)], 0, 2);
        }
    };

    stage(0, 0);
    int buf = 0;
    for (int k0 = 0; k0 < K; k0 += 32, buf ^= 1) {
        __syncthreads();
        if (k0 + 32 < K) stage(k0 + 32, buf ^ 1);  // overlap with compute
        v16bf a0 = load_a_frag(&sA[buf][(wm + 0)  * 32], 32);
        v16bf a1 = load_a_frag(&sA[buf][(wm + 16) * 32], 32);
#pragma unroll
        for (int j = 0; j < 4; ++j) {
            v16bf bj = load_b_frag(&sB[buf][(wn + j * 16) * 32], 32);
            acc[0][j] = wmma_bf16(a0, bj, acc[0][j]);
            acc[1][j] = wmma_bf16(a1, bj, acc[1][j]);
        }
    }

    // C/D f32 layout: lane&15 -> column, vgpr r -> row r (+8 for lanes 16..31).
    const int lane = tid & 31;
    const int n    = lane & 15;
    const int hi8  = (lane & 16) ? 8 : 0;
#pragma unroll
    for (int sm = 0; sm < 2; ++sm)
#pragma unroll
        for (int sn = 0; sn < 4; ++sn)
#pragma unroll
            for (int r = 0; r < 8; ++r) {
                int row = m0 + wm + sm * 16 + r + hi8;
                int col = n0 + wn + sn * 16 + n;
                C[(size_t)row * N + col] = (TO)acc[sm][sn][r];
            }
}

// ---------------------------------------------------------------------------
// Flash attention + fused per-head PMSNorm.
// Grid: B*H*(S/64) blocks, 128 threads = 4 waves; wave w handles q-rows
// [q0+16w, q0+16w+16). K/V tiles of 32 keys staged in LDS (V transposed).
// ---------------------------------------------------------------------------
__global__ __launch_bounds__(128)
void attn_kernel(const bf16_t* __restrict__ qkv,
                 const float* __restrict__ norm_w,
                 bf16_t* __restrict__ attn_out) {
    __shared__ __align__(32) bf16_t kT[32 * Dd];      // [key][d]   16 KB
    __shared__ __align__(32) bf16_t vT[Dd * 32];      // [d][key]   16 KB
    __shared__ __align__(32) bf16_t pbuf[4][16 * 32]; // per-wave P  4 KB

    const int tid  = threadIdx.x;
    const int lane = tid & 31;
    const int wave = tid >> 5;
    const int n    = lane & 15;
    const int hi8  = (lane & 16) ? 8 : 0;

    const int qb = blockIdx.x & (Ss / 64 - 1);
    const int h  = (blockIdx.x >> 5) & (Hh - 1);
    const int b  = blockIdx.x >> 8;
    const int q0  = qb * 64;
    const int qr0 = q0 + wave * 16;    // this wave's first q row

    // Q fragments for this wave's 16 rows: 8 frags cover d=0..255.
    const bf16_t* qbase = qkv + ((size_t)(b * Ss + qr0)) * QKV_LD + h * Dd;
    v16bf qf[8];
#pragma unroll
    for (int f = 0; f < 8; ++f) qf[f] = load_a_frag(qbase + f * 32, QKV_LD);

    v8f   acc[16] = {};           // O tile 16 x 256 (16 n-tiles)
    float m[8], l[8];
#pragma unroll
    for (int r = 0; r < 8; ++r) { m[r] = NEGf; l[r] = 0.0f; }

    const float sc = 1.0f / 64.0f;   // 1/(sqrt(256)*(layer_idx+1))

    for (int j0 = 0; j0 < q0 + 64; j0 += 32) {
        __syncthreads();
        // Stage K (row-major, async copy to LDS) and V (transposed via VGPRs).
        const bf16_t* kb = qkv + ((size_t)(b * Ss + j0)) * QKV_LD + Ee + h * Dd;
        const bf16_t* vb = qkv + ((size_t)(b * Ss + j0)) * QKV_LD + 2 * Ee + h * Dd;
#pragma unroll
        for (int i = 0; i < 8; ++i) {
            int e8  = tid + i * 128;     // 1024 chunks of 8 bf16 (16B)
            int row = e8 >> 5;
            int col = (e8 & 31) * 8;
#ifdef HAVE_ASYNC_LDS
            __builtin_amdgcn_global_load_async_to_lds_b128(
                (__attribute__((address_space(1))) v4i_gcc*)
                    (kb + (size_t)row * QKV_LD + col),
                (__attribute__((address_space(3))) v4i_gcc*)
                    &kT[row * Dd + col],
                0, 0);
#else
            *(v8bf*)&kT[row * Dd + col] =
                *(const v8bf*)(kb + (size_t)row * QKV_LD + col);
#endif
            v8bf vv = *(const v8bf*)(vb + (size_t)row * QKV_LD + col);
#pragma unroll
            for (int j = 0; j < 8; ++j) vT[(col + j) * 32 + row] = vv[j];
        }
#ifdef HAVE_ASYNC_LDS
        wait_async0();
#endif
        __syncthreads();

        if (j0 <= qr0 + 15) {          // wave has unmasked work this step
            // scores: S = Q (16x256) * K^T (256x32)  -> two 16x16 tiles
            v8f s0 = {}, s1 = {};
#pragma unroll
            for (int f = 0; f < 8; ++f) {
                v16bf bk0 = load_b_frag(&kT[0  * Dd + f * 32], Dd);
                v16bf bk1 = load_b_frag(&kT[16 * Dd + f * 32], Dd);
                s0 = wmma_bf16(qf[f], bk0, s0);
                s1 = wmma_bf16(qf[f], bk1, s1);
            }
            // scale + causal mask + online softmax (row lives in 16 lanes)
            float p0[8], p1[8], corr[8];
#pragma unroll
            for (int r = 0; r < 8; ++r) {
                int qrow = qr0 + r + hi8;
                float v0 = (j0 + n      <= qrow) ? s0[r] * sc : NEGf;
                float v1 = (j0 + 16 + n <= qrow) ? s1[r] * sc : NEGf;
                float mx = fmaxf(v0, v1);
                mx = fmaxf(mx, __shfl_xor(mx, 1));
                mx = fmaxf(mx, __shfl_xor(mx, 2));
                mx = fmaxf(mx, __shfl_xor(mx, 4));
                mx = fmaxf(mx, __shfl_xor(mx, 8));
                float mnew = fmaxf(m[r], mx);
                float e0 = __expf(v0 - mnew);
                float e1 = __expf(v1 - mnew);
                p0[r] = e0; p1[r] = e1;
                float ps = e0 + e1;
                ps += __shfl_xor(ps, 1);
                ps += __shfl_xor(ps, 2);
                ps += __shfl_xor(ps, 4);
                ps += __shfl_xor(ps, 8);
                corr[r] = __expf(m[r] - mnew);
                l[r] = l[r] * corr[r] + ps;
                m[r] = mnew;
            }
#pragma unroll
            for (int t = 0; t < 16; ++t)
#pragma unroll
                for (int r = 0; r < 8; ++r) acc[t][r] *= corr[r];

            // Repack P into A-layout via per-wave LDS buffer.
            bf16_t* pb = &pbuf[wave][0];
#pragma unroll
            for (int r = 0; r < 8; ++r) {
                pb[(r + hi8) * 32 + n]      = (bf16_t)p0[r];
                pb[(r + hi8) * 32 + 16 + n] = (bf16_t)p1[r];
            }
            asm volatile("s_wait_dscnt 0" ::: "memory");  // wave-local LDS RAW
            v16bf pa = load_a_frag(pb, 32);
            // O += P (16x32) * V (32x256)
#pragma unroll
            for (int t = 0; t < 16; ++t) {
                v16bf bv = load_b_frag(&vT[t * 16 * 32], 32);
                acc[t] = wmma_bf16(pa, bv, acc[t]);
            }
        }
    }

    // Epilogue: 1/l, PMSNorm over D=256, norm weight, store merged-head bf16.
    float inv_l[8], rnorm[8];
#pragma unroll
    for (int r = 0; r < 8; ++r) inv_l[r] = 1.0f / l[r];
#pragma unroll
    for (int r = 0; r < 8; ++r) {
        float ss = 0.0f;
#pragma unroll
        for (int t = 0; t < 16; ++t) {
            float o = acc[t][r] * inv_l[r];
            ss += o * o;
        }
        ss += __shfl_xor(ss, 1);
        ss += __shfl_xor(ss, 2);
        ss += __shfl_xor(ss, 4);
        ss += __shfl_xor(ss, 8);
        rnorm[r] = rsqrtf(ss * (1.0f / (float)Dd) + EPSf);
    }
    bf16_t* ob = attn_out + ((size_t)(b * Ss + qr0)) * Ee + h * Dd;
#pragma unroll
    for (int t = 0; t < 16; ++t) {
        float nw = norm_w[t * 16 + n];
#pragma unroll
        for (int r = 0; r < 8; ++r) {
            float o = acc[t][r] * inv_l[r] * rnorm[r] * nw;
            ob[(size_t)(r + hi8) * Ee + t * 16 + n] = (bf16_t)o;
        }
    }
}

// ---------------------------------------------------------------------------
extern "C" void kernel_launch(void* const* d_in, const int* in_sizes, int n_in,
                              void* d_out, int out_size, void* d_ws, size_t ws_size,
                              hipStream_t stream) {
    (void)in_sizes; (void)n_in; (void)out_size; (void)ws_size;
    const float* hidden = (const float*)d_in[0];   // [B,S,E]
    const float* w_attn = (const float*)d_in[1];   // [E,3E]
    const float* w_proj = (const float*)d_in[2];   // [E,E]
    const float* norm_w = (const float*)d_in[3];   // [D]
    float* out = (float*)d_out;                    // [B,S,E] f32

    bf16_t* qkv    = (bf16_t*)d_ws;                       // [B*S, 3E] bf16
    bf16_t* attn_o = qkv + (size_t)(Bb * Ss) * (3 * Ee);  // [B*S, E]  bf16

    const int M = Bb * Ss;  // 4096

    // 1) QKV projection: f32 in -> bf16 out
    {
        dim3 grid((3 * Ee) / 128, M / 128);
        gemm_bf16<float, bf16_t><<<grid, 256, 0, stream>>>(
            hidden, w_attn, qkv, M, 3 * Ee, Ee);
    }
    // 2) Flash attention + PMSNorm: bf16 in -> bf16 out (heads merged)
    {
        dim3 grid(Bb * Hh * (Ss / 64));
        attn_kernel<<<grid, 128, 0, stream>>>(qkv, norm_w, attn_o);
    }
    // 3) Output projection: bf16 in -> f32 out
    {
        dim3 grid(Ee / 128, M / 128);
        gemm_bf16<bf16_t, float><<<grid, 256, 0, stream>>>(
            attn_o, w_proj, out, M, Ee, Ee);
    }
}
